// GAT_hetero_25323127177269
// MI455X (gfx1250) — matrix operand order repaired and checked by reading, not compiled, not run on hardware
//
#include <hip/hip_runtime.h>

// ---------------------------------------------------------------------------
// Hetero-GAT (4 layers x 2 edge types) for gfx1250 (MI455X, wave32, WMMA).
//   N=100000 nodes, E_int=1.6M, E_ext=0.4M edges.
//   Layer (fin, D, H): (64,32,4) (128,16,4) (64,8,4) (32,1,1)
// GEMMs: v_wmma_f32_16x16x32_f16 with pre-packed f16 operands:
//   A = xh[N][fin] (f16), B = Wt[HDp][fin] (f16, transposed + zero-padded)
//   -> inner loop is 4x global_load_b128 + 1x v_wmma, no branches, no cvts.
// Edge aggregation: one wave32 per edge, lanes sweep H*D features (coalesced),
// alpha broadcast via __shfl; h / accumulator are <=51MB -> L2-resident.
// ---------------------------------------------------------------------------

#define NNODES 100000
#define NEG_SLOPE 0.2f

typedef __attribute__((ext_vector_type(16))) _Float16 v16h;
typedef __attribute__((ext_vector_type(8)))  _Float16 v8h;
typedef __attribute__((ext_vector_type(8)))  float    v8f;

__device__ __forceinline__ void atomicMaxFloat(float* addr, float v) {
    // Ordered-int trick (addr initialized to -inf).
    if (v >= 0.0f) atomicMax((int*)addr, __float_as_int(v));
    else           atomicMin((unsigned int*)addr, __float_as_uint(v));
}

// ---------------- operand packing (once per layer / etype) -----------------
__global__ void pack_x_f16(const float* __restrict__ x, _Float16* __restrict__ xh,
                           int n) {
    int i = blockIdx.x * blockDim.x + threadIdx.x;
    if (i < n) xh[i] = (_Float16)x[i];
}

// Wt[n*fin + k] = (n < HD) ? W[k*HD + n] : 0   for n in [0, HDp)
__global__ void pack_w_tr(const float* __restrict__ W, _Float16* __restrict__ Wt,
                          int fin, int HD, int HDp) {
    int i = blockIdx.x * blockDim.x + threadIdx.x;
    if (i >= fin * HDp) return;
    int n = i / fin, k = i - n * fin;
    Wt[(size_t)n * fin + k] = (n < HD) ? (_Float16)W[(size_t)k * HD + n]
                                       : (_Float16)0.0f;
}

// ------------------------- h = x @ W  (WMMA) -------------------------------
// One wave32 per 16x16 output tile. K stepped by 32 (fin in {32,64,128}).
// A 16x32 f16 layout: lanes 0-15: M=lane, halves 0..7=K 0..7, 8..15=K 16..23;
//                     lanes 16-31: M=lane-16, K bases +8.
// B 32x16 f16 layout: same K order as A (lanes 0-15: K 0..7,16..23; lanes 16-31: K 8..15,24..31).
// D f32 layout: VGPR r -> M = r (+8 for lanes 16-31), N = lane&15.
__global__ void gemm_wmma_f16(const _Float16* __restrict__ xh,
                              const _Float16* __restrict__ Wt,
                              float* __restrict__ h,
                              int fin, int HD, int ntn, int ntiles) {
    int wave = (int)((blockIdx.x * blockDim.x + threadIdx.x) >> 5);
    int lane = threadIdx.x & 31;
    if (wave >= ntiles) return;
    int mt = wave / ntn;
    int nt = wave % ntn;

    int row  = mt * 16 + (lane & 15);
    int ncol = nt * 16 + (lane & 15);
    int ka   = (lane < 16) ? 0 : 8;     // A K-base: halves 0..7 -> ka.., 8..15 -> ka+16..
    int kb   = (lane < 16) ? 0 : 8;     // B K-base: halves 0..7 -> kb.., 8..15 -> kb+16.. (same as A)

    const _Float16* xr = xh + (size_t)row  * fin;
    const _Float16* wr = Wt + (size_t)ncol * fin;

    v8f c = {};
    for (int k0 = 0; k0 < fin; k0 += 32) {
        v8h a0 = *(const v8h*)(xr + k0 + ka);
        v8h a1 = *(const v8h*)(xr + k0 + ka + 16);
        v8h b0 = *(const v8h*)(wr + k0 + kb);
        v8h b1 = *(const v8h*)(wr + k0 + kb + 16);
        v16h a = __builtin_shufflevector(a0, a1, 0,1,2,3,4,5,6,7,8,9,10,11,12,13,14,15);
        v16h b = __builtin_shufflevector(b0, b1, 0,1,2,3,4,5,6,7,8,9,10,11,12,13,14,15);
        // (neg_a, A, neg_b, B, c_mod, C, reuse_a, reuse_b)
        c = __builtin_amdgcn_wmma_f32_16x16x32_f16(false, a, false, b,
                                                   (short)0, c, false, false);
    }

    if (ncol < HD) {
        int m0 = mt * 16 + ((lane < 16) ? 0 : 8);
#pragma unroll
        for (int r = 0; r < 8; ++r)
            h[(size_t)(m0 + r) * HD + ncol] = c[r];
    }
}

// -------- per-(node,head): el/er dot products + init segment max/sum -------
__global__ void attn_coef(const float* __restrict__ h,
                          const float* __restrict__ al,
                          const float* __restrict__ ar,
                          float* __restrict__ el, float* __restrict__ er,
                          float* __restrict__ mx, float* __restrict__ sm,
                          int H, int D, int total) {
    int i = blockIdx.x * blockDim.x + threadIdx.x;   // node*H + head
    if (i >= total) return;
    int head = i % H;
    const float* hp  = h + (size_t)(i / H) * (H * D) + head * D;
    const float* alp = al + head * D;
    const float* arp = ar + head * D;
    float a = 0.0f, b = 0.0f;
    for (int d = 0; d < D; ++d) { float v = hp[d]; a += v * alp[d]; b += v * arp[d]; }
    el[i] = a; er[i] = b;
    mx[i] = -__builtin_inff();
    sm[i] = 0.0f;
}

__global__ void zero_f32(float* __restrict__ p, int n) {
    int i = blockIdx.x * blockDim.x + threadIdx.x;
    if (i < n) p[i] = 0.0f;
}

// ------------- edge pass 1: segment max (1 thread/edge, loop heads) --------
__global__ void edge_max(const int* __restrict__ src, const int* __restrict__ dst,
                         const float* __restrict__ el, const float* __restrict__ er,
                         float* __restrict__ mx, int E, int H) {
    int e = blockIdx.x * blockDim.x + threadIdx.x;
    if (e >= E) return;
    int s = src[e], d = dst[e];
    for (int hh = 0; hh < H; ++hh) {
        float v = el[s * H + hh] + er[d * H + hh];
        v = (v > 0.0f) ? v : v * NEG_SLOPE;
        atomicMaxFloat(&mx[d * H + hh], v);
    }
}

// ------------- edge pass 2: exp-sum (1 thread/edge, loop heads) ------------
__global__ void edge_sum(const int* __restrict__ src, const int* __restrict__ dst,
                         const float* __restrict__ el, const float* __restrict__ er,
                         const float* __restrict__ mx, float* __restrict__ sm,
                         int E, int H) {
    int e = blockIdx.x * blockDim.x + threadIdx.x;
    if (e >= E) return;
    int s = src[e], d = dst[e];
    for (int hh = 0; hh < H; ++hh) {
        float v = el[s * H + hh] + er[d * H + hh];
        v = (v > 0.0f) ? v : v * NEG_SLOPE;
        atomicAdd(&sm[d * H + hh], __expf(v - mx[d * H + hh]));
    }
}

// ------------- edge pass 3: weighted scatter-aggregate (wave/edge) ---------
__global__ void edge_agg(const int* __restrict__ src, const int* __restrict__ dst,
                         const float* __restrict__ h,
                         const float* __restrict__ el, const float* __restrict__ er,
                         const float* __restrict__ mx, const float* __restrict__ sm,
                         float* __restrict__ acc, int E, int H, int D) {
    int wave = (int)((blockIdx.x * blockDim.x + threadIdx.x) >> 5);
    int lane = threadIdx.x & 31;
    if (wave >= E) return;
    int s = src[wave], d = dst[wave];

    float alpha = 0.0f;
    if (lane < H) {
        float v = el[s * H + lane] + er[d * H + lane];
        v = (v > 0.0f) ? v : v * NEG_SLOPE;
        alpha = __expf(v - mx[d * H + lane]) / sm[d * H + lane];
    }
    int HD = H * D;
    const float* hs = h + (size_t)s * HD;
    float* ad = acc + (size_t)d * HD;
    for (int f = lane; f < HD; f += 32) {
        float af = __shfl(alpha, f / D, 32);   // broadcast head's alpha
        atomicAdd(&ad[f], hs[f] * af);
    }
}

// --- inner layers: add both biases, ReLU, emit f16 features for next GEMM --
__global__ void finalize_f16(const float* __restrict__ acc,
                             const float* __restrict__ b0, const float* __restrict__ b1,
                             _Float16* __restrict__ xh, int total, int HD) {
    int i = blockIdx.x * blockDim.x + threadIdx.x;
    if (i >= total) return;
    int f = i % HD;
    float v = acc[i] + b0[f] + b1[f];
    v = (v > 0.0f) ? v : 0.0f;
    xh[i] = (_Float16)v;
}

// --- last layer: add both biases, write f32 output ([N,1]) -----------------
__global__ void finalize_out(const float* __restrict__ acc,
                             const float* __restrict__ b0, const float* __restrict__ b1,
                             float* __restrict__ out, int total) {
    int i = blockIdx.x * blockDim.x + threadIdx.x;
    if (i >= total) return;
    out[i] = acc[i] + b0[0] + b1[0];
}

// ---------------------------------------------------------------------------
extern "C" void kernel_launch(void* const* d_in, const int* in_sizes, int n_in,
                              void* d_out, int out_size, void* d_ws, size_t ws_size,
                              hipStream_t stream) {
    const float* feat = (const float*)d_in[0];
    const int* esrc[2] = { (const int*)d_in[1], (const int*)d_in[3] };  // interacts, external
    const int* edst[2] = { (const int*)d_in[2], (const int*)d_in[4] };
    const int  ecnt[2] = { 1600000, 400000 };

    // params flattened: t in {interacts, external} x layer x (W, al, ar, b)
    auto P = [&](int t, int l, int w) -> const float* {
        return (const float*)d_in[5 + t * 16 + l * 4 + w];
    };

    const int fin_a[4]  = { 64, 128, 64, 32 };
    const int dout_a[4] = { 32, 16, 8, 1 };
    const int H_a[4]    = { 4, 4, 4, 1 };

    // ---- workspace layout: f32 region, then f16 region (16B aligned) -----
    float* ws = (float*)d_ws;
    size_t off = 0;
    float* hbuf[2]; hbuf[0] = ws + off; off += (size_t)NNODES * 128;
    hbuf[1] = ws + off; off += (size_t)NNODES * 128;
    float* acc = ws + off; off += (size_t)NNODES * 128;
    float *el[2], *er[2], *mx[2], *sm[2];
    for (int t = 0; t < 2; ++t) {
        el[t] = ws + off; off += (size_t)NNODES * 4;
        er[t] = ws + off; off += (size_t)NNODES * 4;
        mx[t] = ws + off; off += (size_t)NNODES * 4;
        sm[t] = ws + off; off += (size_t)NNODES * 4;
    }
    _Float16* xh = (_Float16*)(ws + off);              // N*128 halves
    _Float16* wh[2];
    wh[0] = xh + (size_t)NNODES * 128;                 // 128*128 halves each
    wh[1] = wh[0] + 128 * 128;

    const int TB = 256;

    // layer 0 input: pack feat -> f16 once
    {
        int n = NNODES * 64;
        pack_x_f16<<<(n + TB - 1) / TB, TB, 0, stream>>>(feat, xh, n);
    }

    for (int l = 0; l < 4; ++l) {
        int fin = fin_a[l], D = dout_a[l], H = H_a[l];
        int HD = H * D;
        int ntn = (HD + 15) / 16;
        int HDp = ntn * 16;
        int ntiles = (NNODES / 16) * ntn;

        for (int t = 0; t < 2; ++t) {
            // pack W -> transposed, zero-padded f16
            int wn = fin * HDp;
            pack_w_tr<<<(wn + TB - 1) / TB, TB, 0, stream>>>(P(t, l, 0), wh[t],
                                                             fin, HD, HDp);
            // projection h_t = x @ W_t  (WMMA)
            int gblk = (ntiles + 7) / 8;   // 8 waves per 256-thread block
            gemm_wmma_f16<<<gblk, TB, 0, stream>>>(xh, wh[t], hbuf[t],
                                                   fin, HD, ntn, ntiles);
            // el/er + init segment max/sum
            int tot = NNODES * H;
            attn_coef<<<(tot + TB - 1) / TB, TB, 0, stream>>>(
                hbuf[t], P(t, l, 1), P(t, l, 2), el[t], er[t], mx[t], sm[t],
                H, D, tot);
        }

        int totHD = NNODES * HD;
        zero_f32<<<(totHD + TB - 1) / TB, TB, 0, stream>>>(acc, totHD);

        for (int t = 0; t < 2; ++t) {
            int E = ecnt[t];
            edge_max<<<(E + TB - 1) / TB, TB, 0, stream>>>(
                esrc[t], edst[t], el[t], er[t], mx[t], E, H);
            edge_sum<<<(E + TB - 1) / TB, TB, 0, stream>>>(
                esrc[t], edst[t], el[t], er[t], mx[t], sm[t], E, H);
            edge_agg<<<(E + 7) / 8, TB, 0, stream>>>(
                esrc[t], edst[t], hbuf[t], el[t], er[t], mx[t], sm[t],
                acc, E, H, D);
        }

        if (l < 3) {
            finalize_f16<<<(totHD + TB - 1) / TB, TB, 0, stream>>>(
                acc, P(0, l, 3), P(1, l, 3), xh, totHD, HD);
        } else {
            finalize_out<<<(NNODES + TB - 1) / TB, TB, 0, stream>>>(
                acc, P(0, l, 3), P(1, l, 3), (float*)d_out, NNODES);
        }
    }
}
